// Model_41274635714652
// MI455X (gfx1250) — matrix operand order repaired and checked by reading, not compiled, not run on hardware
//
#include <hip/hip_runtime.h>
#include <hip/hip_bf16.h>

// ---------------------------------------------------------------------------
// PointMLP-ish forward for gfx1250 (MI455X).  GEMM-heavy parts use
// v_wmma_f32_16x16x32_bf16 with double-buffered LDS tiles staged in WMMA
// fragment order (fragment loads = ds_load_b128, batched before the WMMA
// group so DS waits are graduated) and branch-free clamped global loads.
// All reductions fixed-order (deterministic).
// ---------------------------------------------------------------------------

typedef __attribute__((ext_vector_type(16))) __bf16 v16bf;
typedef __attribute__((ext_vector_type(2)))  __bf16 bf16x2;
typedef __attribute__((ext_vector_type(8)))  float  v8f;

#define BATCH 8
#define NPTS0 5000
#define KNN_K 32

#define BM 64
#define BNT 128
#define KT 32

static __device__ __forceinline__ bf16x2 pack_bf16(float v0, float v1)
{
#if __has_builtin(__builtin_amdgcn_cvt_pk_bf16_f32)
    return __builtin_amdgcn_cvt_pk_bf16_f32(v0, v1);   // single v_cvt_pk_bf16_f32
#else
    return (bf16x2){(__bf16)v0, (__bf16)v1};
#endif
}

// Out[g*M*GS + m*GS + p] = sum_k W[m*Kd+k]*X[(g*Kd+k)*GS+p] + bias[m]
// column l in [0,L): g = l/GS, p = l%GS.  Covers conv1d [B,C,N] (GS=N),
// conv "4d" [B*S, C, K] (GS=K) and dense layers (GS=1).
__global__ __launch_bounds__(256)
void gemm_bf16_wmma(const float* __restrict__ W, const float* __restrict__ bias,
                    const float* __restrict__ X, float* __restrict__ Out,
                    int M, int Kd, long long GS, long long L)
{
    // double-buffered LDS tiles in WMMA fragment order: [buf][tile16][lane][elem]
    __shared__ __align__(16) __bf16 sWf[2][BM / 16][32][16];
    __shared__ __align__(16) __bf16 sXf[2][BNT / 16][32][16];
    __shared__ long long colBase[BNT];   // g*Kd*GS + p per column (-1 = OOB)

    const int tid  = threadIdx.x;
    const int lane = tid & 31;
    const int wave = tid >> 5;
    const int wm   = wave & 3;     // M sub-tile (16 rows)
    const int wn   = wave >> 2;    // 0..1 -> 64-column half
    const long long nb = (long long)blockIdx.x * BNT;
    const int       mb = blockIdx.y * BM;

    // one-time per-column address bases (hoists the 64-bit divide)
    if (tid < BNT) {
        long long l = nb + tid;
        long long cb = -1;
        if (l < L) {
            long long g = l / GS, p = l - g * GS;
            cb = g * Kd * GS + p;
        }
        colBase[tid] = cb;
    }

    v8f acc[4];
#pragma unroll
    for (int t = 0; t < 4; ++t)
#pragma unroll
        for (int i = 0; i < 8; ++i) acc[t][i] = 0.f;

    __syncthreads();   // colBase visible to all staging threads

    // ---- branch-free staging (clamped addresses, selected values) ----
    auto stageW = [&](int kt, int buf) {
#pragma unroll
        for (int i = tid; i < BM * (KT / 2); i += 256) {
            int mm = i >> 4;
            int kk = (i & 15) << 1;            // even k within tile
            int m = mb + mm, k = kt * KT + kk;
            bool ok0 = (m < M) && (k < Kd);
            bool ok1 = (m < M) && (k + 1 < Kd);
            long long a0 = ok0 ? ((long long)m * Kd + k)     : 0;
            long long a1 = ok1 ? ((long long)m * Kd + k + 1) : 0;
            float v0 = W[a0];
            float v1 = W[a1];
            v0 = ok0 ? v0 : 0.f;
            v1 = ok1 ? v1 : 0.f;
            // A layout: lane = ((k>>3)&1)*16 + m%16 ; elem = ((k>>4)<<3)+(k&7)
            int ln = (((kk >> 3) & 1) << 4) | (mm & 15);
            int e  = ((kk >> 4) << 3) | (kk & 7);
            *(bf16x2*)&sWf[buf][mm >> 4][ln][e] = pack_bf16(v0, v1);
        }
    };
    auto stageX = [&](int kt, int buf) {
#pragma unroll
        for (int i = tid; i < BNT * (KT / 2); i += 256) {
            int nn = i >> 4;
            int kk = (i & 15) << 1;
            int k = kt * KT + kk;
            long long cb = colBase[nn];
            bool ok0 = (cb >= 0) && (k < Kd);
            bool ok1 = (cb >= 0) && (k + 1 < Kd);
            long long a0 = ok0 ? cb + (long long)k * GS       : 0;
            long long a1 = ok1 ? cb + (long long)(k + 1) * GS : 0;
            float v0 = X[a0];
            float v1 = X[a1];
            v0 = ok0 ? v0 : 0.f;
            v1 = ok1 ? v1 : 0.f;
            // B layout: lane = ((k>>3)&1)*16 + n%16 ; elem = ((k>>4)<<3)+(k&7)
            int ln = (((kk >> 3) & 1) << 4) | (nn & 15);
            int e  = ((kk >> 4) << 3) | (kk & 7);
            *(bf16x2*)&sXf[buf][nn >> 4][ln][e] = pack_bf16(v0, v1);
        }
    };

    const int nkt = (Kd + KT - 1) / KT;
    stageW(0, 0);
    stageX(0, 0);
    __syncthreads();

    for (int kt = 0; kt < nkt; ++kt) {
        const int cur = kt & 1, nxt = cur ^ 1;
        // stage next tile while computing this one (loads overlap WMMA)
        if (kt + 1 < nkt) {
            stageW(kt + 1, nxt);
            stageX(kt + 1, nxt);
        }
        // speculative prefetch two tiles ahead (global_prefetch_b8)
        if (kt + 2 < nkt) {
            long long cb = colBase[tid & (BNT - 1)];
            int k = (kt + 2) * KT + (tid >> 7);
            if (cb >= 0 && k < Kd)
                __builtin_prefetch(&X[cb + (long long)k * GS], 0, 1);
        }
        // ---- batch all fragment loads (ds_load_b128 x10), then 4 WMMAs ----
        const v16bf a = *(const v16bf*)sWf[cur][wm][lane];
        v16bf bf[4];
#pragma unroll
        for (int t = 0; t < 4; ++t)
            bf[t] = *(const v16bf*)sXf[cur][(wn << 2) + t][lane];
#pragma unroll
        for (int t = 0; t < 4; ++t)
            acc[t] = __builtin_amdgcn_wmma_f32_16x16x32_bf16(
                false, a, false, bf[t], (short)0, acc[t], false, false);
        __syncthreads();   // single barrier per K-step (double buffer)
    }

    // ---- epilogue: C layout row = (lane/16)*8 + v, col = lane%16 ----
    const int rbase = mb + wm * 16 + ((lane >> 4) << 3);
    float bv[8];
#pragma unroll
    for (int v = 0; v < 8; ++v)
        bv[v] = (bias && rbase + v < M) ? bias[rbase + v] : 0.f;
#pragma unroll
    for (int t = 0; t < 4; ++t) {
        long long l = nb + (wn << 6) + (t << 4) + (lane & 15);
        if (l >= L) continue;
        long long g = l / GS, p = l - g * GS;
        long long ob = g * M * GS + p;
#pragma unroll
        for (int v = 0; v < 8; ++v) {
            int m = rbase + v;
            if (m < M)
                Out[ob + (long long)m * GS] = acc[t][v] + bv[v];
        }
    }
}

// ---- training-mode BatchNorm: per-channel mean/invstd over (groups, GS) ----
__global__ __launch_bounds__(256)
void bn_stats(const float* __restrict__ X, float* __restrict__ stats,
              int M, long long GS, long long nG, float eps)
{
    const int m = blockIdx.x;
    const long long cnt = GS * nG;
    __shared__ float ss[256], sq[256];
    float s = 0.f, q = 0.f;
    for (long long i = threadIdx.x; i < cnt; i += 256) {
        long long g = i / GS, p = i - g * GS;
        float v = X[(g * M + m) * GS + p];
        s += v; q += v * v;
    }
    ss[threadIdx.x] = s; sq[threadIdx.x] = q;
    __syncthreads();
    for (int o = 128; o > 0; o >>= 1) {
        if (threadIdx.x < o) {
            ss[threadIdx.x] += ss[threadIdx.x + o];
            sq[threadIdx.x] += sq[threadIdx.x + o];
        }
        __syncthreads();
    }
    if (threadIdx.x == 0) {
        float mean = ss[0] / (float)cnt;
        float var  = sq[0] / (float)cnt - mean * mean;
        stats[m]     = mean;
        stats[M + m] = rsqrtf(fmaxf(var, 0.f) + eps);
    }
}

__global__ __launch_bounds__(256)
void bn_apply(float* __restrict__ X, const float* __restrict__ stats,
              const float* __restrict__ res, int M, long long GS,
              long long total, int relu)
{
    const long long MG = (long long)M * GS;
    for (long long i = (long long)blockIdx.x * 256 + threadIdx.x; i < total;
         i += (long long)gridDim.x * 256) {
        long long r = i % MG;
        int m = (int)(r / GS);
        float v = (X[i] - stats[m]) * stats[M + m];
        if (res)  v += res[i];
        if (relu) v = fmaxf(v, 0.f);
        X[i] = v;
    }
}

// ---- farthest point sampling: one block per batch, sequential over S ----
__global__ __launch_bounds__(1024)
void fps_kernel(const float* __restrict__ xyz, int* __restrict__ out, int Np, int S)
{
    const int b = blockIdx.x;
    const float* P = xyz + (long long)b * Np * 5;
    __shared__ float dmin[5024];
    __shared__ float rv[1024];
    __shared__ int   ri[1024];
    __shared__ int   sFar;
    for (int i = threadIdx.x; i < Np; i += 1024) dmin[i] = 1e10f;
    if (threadIdx.x == 0) sFar = 0;
    __syncthreads();
    for (int s = 0; s < S; ++s) {
        const int far = sFar;
        if (threadIdx.x == 0) out[b * S + s] = far;
        const float c0 = P[far*5+0], c1 = P[far*5+1], c2 = P[far*5+2],
                    c3 = P[far*5+3], c4 = P[far*5+4];
        float bv = -1e30f; int bi = 0x7fffffff;
        for (int i = threadIdx.x; i < Np; i += 1024) {
            float d0 = P[i*5+0]-c0, d1 = P[i*5+1]-c1, d2 = P[i*5+2]-c2,
                  d3 = P[i*5+3]-c3, d4 = P[i*5+4]-c4;
            float d = d0*d0 + d1*d1 + d2*d2 + d3*d3 + d4*d4;
            float dm = fminf(dmin[i], d);
            dmin[i] = dm;
            if (dm > bv || (dm == bv && i < bi)) { bv = dm; bi = i; }
        }
        rv[threadIdx.x] = bv; ri[threadIdx.x] = bi;
        __syncthreads();
        for (int o = 512; o > 0; o >>= 1) {
            if (threadIdx.x < o) {
                float v2 = rv[threadIdx.x + o]; int i2 = ri[threadIdx.x + o];
                if (v2 > rv[threadIdx.x] ||
                    (v2 == rv[threadIdx.x] && i2 < ri[threadIdx.x])) {
                    rv[threadIdx.x] = v2; ri[threadIdx.x] = i2;
                }
            }
            __syncthreads();
        }
        if (threadIdx.x == 0) sFar = ri[0];
        __syncthreads();
    }
}

// ---- kNN (32 smallest, ascending, lowest-index tie-break) ----
__global__ __launch_bounds__(256)
void knn_kernel(const float* __restrict__ xyz, const float* __restrict__ nxyz,
                int* __restrict__ out, int Np, int S)
{
    const int b = blockIdx.y, s = blockIdx.x;
    __shared__ float dist[5024];
    __shared__ float rv[256];
    __shared__ int   ri[256];
    const float* P = xyz + (long long)b * Np * 5;
    const float* Q = nxyz + ((long long)b * S + s) * 5;
    const float q0 = Q[0], q1 = Q[1], q2 = Q[2], q3 = Q[3], q4 = Q[4];
    for (int i = threadIdx.x; i < Np; i += 256) {
        float d0 = P[i*5+0]-q0, d1 = P[i*5+1]-q1, d2 = P[i*5+2]-q2,
              d3 = P[i*5+3]-q3, d4 = P[i*5+4]-q4;
        dist[i] = d0*d0 + d1*d1 + d2*d2 + d3*d3 + d4*d4;
    }
    __syncthreads();
    for (int kk = 0; kk < KNN_K; ++kk) {
        float bv = 3.3e38f; int bi = 0x7fffffff;
        for (int i = threadIdx.x; i < Np; i += 256) {
            float d = dist[i];
            if (d < bv || (d == bv && i < bi)) { bv = d; bi = i; }
        }
        rv[threadIdx.x] = bv; ri[threadIdx.x] = bi;
        __syncthreads();
        for (int o = 128; o > 0; o >>= 1) {
            if (threadIdx.x < o) {
                float v2 = rv[threadIdx.x + o]; int i2 = ri[threadIdx.x + o];
                if (v2 < rv[threadIdx.x] ||
                    (v2 == rv[threadIdx.x] && i2 < ri[threadIdx.x])) {
                    rv[threadIdx.x] = v2; ri[threadIdx.x] = i2;
                }
            }
            __syncthreads();
        }
        if (threadIdx.x == 0) {
            out[((long long)b * S + s) * KNN_K + kk] = ri[0];
            dist[ri[0]] = 3.4e38f;
        }
        __syncthreads();
    }
}

// ---- misc elementwise kernels ----
__global__ void transpose_cn_kernel(const float* __restrict__ src, float* __restrict__ dst,
                                    int C, int Nn, long long total)
{   // src [B,C,Nn] -> dst [B,Nn,C]
    for (long long i = (long long)blockIdx.x * 256 + threadIdx.x; i < total;
         i += (long long)gridDim.x * 256) {
        int c = (int)(i % C);
        long long bn = i / C;
        int n = (int)(bn % Nn);
        long long b = bn / Nn;
        dst[i] = src[(b * C + c) * Nn + n];
    }
}

__global__ void gather_kernel(const float* __restrict__ src, const int* __restrict__ idx,
                              float* __restrict__ dst, int Np, int S, int C, long long total)
{   // src [B,Np,C], idx [B,S] -> dst [B,S,C]
    for (long long i = (long long)blockIdx.x * 256 + threadIdx.x; i < total;
         i += (long long)gridDim.x * 256) {
        int c = (int)(i % C);
        long long bs = i / C;
        int s = (int)(bs % S);
        long long b = bs / S;
        int p = idx[b * S + s];
        dst[i] = src[(b * Np + p) * C + c];
    }
}

__global__ void group_build_kernel(const float* __restrict__ pts, const float* __restrict__ xyz,
                                   const int* __restrict__ idx, float* __restrict__ G,
                                   int Np, int S, int C, long long total)
{   // G [B,S,K,C+5]
    const int D1 = C + 5;
    for (long long i = (long long)blockIdx.x * 256 + threadIdx.x; i < total;
         i += (long long)gridDim.x * 256) {
        int d = (int)(i % D1);
        long long r = i / D1;
        int k = (int)(r & 31);
        long long bs = r >> 5;
        long long b = bs / S;
        int p = idx[bs * KNN_K + k];
        G[i] = (d < C) ? pts[(b * Np + p) * C + d]
                       : xyz[(b * Np + p) * 5 + (d - C)];
    }
}

__global__ void group_mean_kernel(const float* __restrict__ G, float* __restrict__ mean,
                                  int D1, long long total)
{   // mean [B,S,D1] over K
    for (long long i = (long long)blockIdx.x * 256 + threadIdx.x; i < total;
         i += (long long)gridDim.x * 256) {
        int d = (int)(i % D1);
        long long bs = i / D1;
        float s = 0.f;
        for (int k = 0; k < KNN_K; ++k) s += G[(bs * KNN_K + k) * D1 + d];
        mean[i] = s * (1.f / KNN_K);
    }
}

__global__ __launch_bounds__(1024)
void batch_std_kernel(const float* __restrict__ G, const float* __restrict__ mean,
                      float* __restrict__ stdb, int S, int D1)
{   // one block per batch; std over (S,K,D1), ddof=1
    const int b = blockIdx.x;
    const long long cnt = (long long)S * KNN_K * D1;
    const float* Gb = G + (long long)b * cnt;
    __shared__ float acc[1024];
    float a = 0.f;
    for (long long i = threadIdx.x; i < cnt; i += 1024) {
        int d = (int)(i % D1);
        long long r = i / D1;
        long long s = r >> 5;
        float v = Gb[i] - mean[((long long)b * S + s) * D1 + d];
        a += v * v;
    }
    acc[threadIdx.x] = a;
    __syncthreads();
    for (int o = 512; o > 0; o >>= 1) {
        if (threadIdx.x < o) acc[threadIdx.x] += acc[threadIdx.x + o];
        __syncthreads();
    }
    if (threadIdx.x == 0) stdb[b] = sqrtf(acc[0] / (float)(cnt - 1));
}

__global__ void build_tin_kernel(const float* __restrict__ G, const float* __restrict__ mean,
                                 const float* __restrict__ stdb, const float* __restrict__ alpha,
                                 const float* __restrict__ beta, const float* __restrict__ npts,
                                 float* __restrict__ T, int S, int C, long long total)
{   // T [B,S,D,K], D = 2C+5;  first C+5 dims normalized group, last C dims new_pts
    const int D1 = C + 5, D = 2 * C + 5;
    for (long long i = (long long)blockIdx.x * 256 + threadIdx.x; i < total;
         i += (long long)gridDim.x * 256) {
        int k = (int)(i & 31);
        long long r = i >> 5;
        int d = (int)(r % D);
        long long bs = r / D;
        long long b = bs / S;
        float v;
        if (d < D1) {
            float g = G[(bs * KNN_K + k) * D1 + d];
            float m = mean[bs * D1 + d];
            v = alpha[d] * ((g - m) / (stdb[b] + 1e-5f)) + beta[d];
        } else {
            v = npts[bs * C + (d - D1)];
        }
        T[i] = v;
    }
}

__global__ void maxk_kernel(const float* __restrict__ T, float* __restrict__ h,
                            int S, int C, long long total)
{   // T [B,S,C,K] -> h [B,C,S] (max over K)
    for (long long i = (long long)blockIdx.x * 256 + threadIdx.x; i < total;
         i += (long long)gridDim.x * 256) {
        int s = (int)(i % S);
        long long bc = i / S;
        int c = (int)(bc % C);
        long long b = bc / C;
        const float* p = T + (((b * S + s) * C + c) << 5);
        float m = p[0];
        for (int k = 1; k < KNN_K; ++k) m = fmaxf(m, p[k]);
        h[i] = m;
    }
}

__global__ void maxs_kernel(const float* __restrict__ h, float* __restrict__ feat,
                            int C, int S, long long total)
{   // h [B,C,S] -> feat [B,C]
    for (long long i = (long long)blockIdx.x * 256 + threadIdx.x; i < total;
         i += (long long)gridDim.x * 256) {
        const float* p = h + i * S;
        float m = p[0];
        for (int s = 1; s < S; ++s) m = fmaxf(m, p[s]);
        feat[i] = m;
    }
}

// ---------------------------------------------------------------------------
extern "C" void kernel_launch(void* const* d_in, const int* in_sizes, int n_in,
                              void* d_out, int out_size, void* d_ws, size_t ws_size,
                              hipStream_t stream)
{
    (void)in_sizes; (void)n_in; (void)out_size; (void)ws_size;
    static const int CH[5]  = {64, 128, 256, 512, 1024};
    static const int ANC[4] = {2500, 1250, 625, 312};

    auto fin = [&](int i) { return (const float*)d_in[i]; };
    const float* x      = fin(0);       // [B,5,N]
    const float* emb_w  = fin(1);
    const float* emb_b  = fin(2);
    const float* cls_w1 = fin(83); const float* cls_b1 = fin(84);
    const float* cls_w2 = fin(85); const float* cls_b2 = fin(86);
    const float* cls_w3 = fin(87); const float* cls_b3 = fin(88);

    // ---- workspace carve-up ----
    size_t off = 0;
    auto alloc = [&](size_t bytes) {
        void* p = (char*)d_ws + off;
        off += (bytes + 255) & ~(size_t)255;
        return p;
    };
    const size_t HMAX = 2600000;      // max B*C*S elements
    float* xyzA  = (float*)alloc(sizeof(float) * BATCH * NPTS0 * 5);
    float* xyzB  = (float*)alloc(sizeof(float) * BATCH * NPTS0 * 5);
    float* h0    = (float*)alloc(sizeof(float) * HMAX);
    float* h1    = (float*)alloc(sizeof(float) * HMAX);
    float* s0    = (float*)alloc(sizeof(float) * HMAX);
    float* s1    = (float*)alloc(sizeof(float) * HMAX);
    float* pts   = (float*)alloc(sizeof(float) * HMAX);
    float* npts  = (float*)alloc(sizeof(float) * 1400000);
    int*   fpsIdx = (int*)alloc(sizeof(int) * BATCH * 2500);
    int*   knnIdx = (int*)alloc(sizeof(int) * BATCH * 2500 * KNN_K);
    float* meanB = (float*)alloc(sizeof(float) * 1500000);
    float* stdB  = (float*)alloc(sizeof(float) * 64);
    float* stats = (float*)alloc(sizeof(float) * 2048);
    float* feat  = (float*)alloc(sizeof(float) * BATCH * 1024);
    float* fc1   = (float*)alloc(sizeof(float) * BATCH * 512);
    float* fc2   = (float*)alloc(sizeof(float) * BATCH * 256);
    const size_t BIG = 85200000;      // max B*S*D*K elements
    float* bigA = (float*)alloc(sizeof(float) * BIG);
    float* bigB = (float*)alloc(sizeof(float) * BIG);
    float* bigC = (float*)alloc(sizeof(float) * BIG);

    auto gsz = [](long long total) {
        long long b = (total + 255) / 256;
        if (b > 65535) b = 65535;
        return dim3((unsigned)b);
    };
    auto gemm = [&](const float* Wm, const float* bs, const float* Xm, float* Om,
                    int M, int Kd, long long GS, long long L) {
        dim3 grid((unsigned)((L + BNT - 1) / BNT), (unsigned)((M + BM - 1) / BM));
        gemm_bf16_wmma<<<grid, 256, 0, stream>>>(Wm, bs, Xm, Om, M, Kd, GS, L);
    };
    auto bn = [&](float* Xm, const float* res, int M, long long GS, long long nG, int relu) {
        bn_stats<<<dim3((unsigned)M), 256, 0, stream>>>(Xm, stats, M, GS, nG, 1e-5f);
        long long tot = (long long)M * GS * nG;
        bn_apply<<<gsz(tot), 256, 0, stream>>>(Xm, stats, res, M, GS, tot, relu);
    };

    // ---- xyz = transpose(x) ; embedding = relu(bn(conv(x))) ----
    transpose_cn_kernel<<<gsz((long long)BATCH * NPTS0 * 5), 256, 0, stream>>>(
        x, xyzA, 5, NPTS0, (long long)BATCH * NPTS0 * 5);
    gemm(emb_w, emb_b, x, h0, 64, 5, NPTS0, (long long)BATCH * NPTS0);
    bn(h0, nullptr, 64, NPTS0, BATCH, 1);

    float* curXyz = xyzA;
    float* newXyz = xyzB;
    int Np = NPTS0;

    for (int st = 0; st < 4; ++st) {
        const int cin = CH[st], cout = CH[st + 1];
        const int S = ANC[st];
        const int D1 = cin + 5, D = 2 * cin + 5;
        const int base = 3 + st * 20;
        const float* alpha = fin(base + 0);
        const float* beta  = fin(base + 1);
        const float* tr_w  = fin(base + 2);
        const float* tr_b  = fin(base + 3);
        const float* pre_w1[2] = {fin(base + 4),  fin(base + 8)};
        const float* pre_b1[2] = {fin(base + 5),  fin(base + 9)};
        const float* pre_w2[2] = {fin(base + 6),  fin(base + 10)};
        const float* pre_b2[2] = {fin(base + 7),  fin(base + 11)};
        const float* pos_w1[2] = {fin(base + 12), fin(base + 16)};
        const float* pos_b1[2] = {fin(base + 13), fin(base + 17)};
        const float* pos_w2[2] = {fin(base + 14), fin(base + 18)};
        const float* pos_b2[2] = {fin(base + 15), fin(base + 19)};

        // pts = transpose(h)
        long long tp = (long long)BATCH * Np * cin;
        transpose_cn_kernel<<<gsz(tp), 256, 0, stream>>>(h0, pts, cin, Np, tp);
        // FPS, gathers, kNN
        fps_kernel<<<dim3(BATCH), 1024, 0, stream>>>(curXyz, fpsIdx, Np, S);
        long long tg = (long long)BATCH * S * 5;
        gather_kernel<<<gsz(tg), 256, 0, stream>>>(curXyz, fpsIdx, newXyz, Np, S, 5, tg);
        long long tn = (long long)BATCH * S * cin;
        gather_kernel<<<gsz(tn), 256, 0, stream>>>(pts, fpsIdx, npts, Np, S, cin, tn);
        knn_kernel<<<dim3(S, BATCH), 256, 0, stream>>>(curXyz, newXyz, knnIdx, Np, S);
        // grouping + affine normalization -> t_in [B,S,D,K]
        long long tG = (long long)BATCH * S * KNN_K * D1;
        group_build_kernel<<<gsz(tG), 256, 0, stream>>>(pts, curXyz, knnIdx, bigA, Np, S, cin, tG);
        long long tM = (long long)BATCH * S * D1;
        group_mean_kernel<<<gsz(tM), 256, 0, stream>>>(bigA, meanB, D1, tM);
        batch_std_kernel<<<dim3(BATCH), 1024, 0, stream>>>(bigA, meanB, stdB, S, D1);
        long long tT = (long long)BATCH * S * D * KNN_K;
        build_tin_kernel<<<gsz(tT), 256, 0, stream>>>(bigA, meanB, stdB, alpha, beta,
                                                      npts, bigB, S, cin, tT);
        // PreExtraction: transfer conv + 2 residual blocks on [B*S, C, K]
        const long long GSk = KNN_K;
        const long long Lk  = (long long)BATCH * S * KNN_K;
        const long long nGk = (long long)BATCH * S;
        gemm(tr_w, tr_b, bigB, bigC, cout, D, GSk, Lk);
        bn(bigC, nullptr, cout, GSk, nGk, 1);
        // block 0: bigC -> bigA -> bigB (+res bigC)
        gemm(pre_w1[0], pre_b1[0], bigC, bigA, cout, cout, GSk, Lk);
        bn(bigA, nullptr, cout, GSk, nGk, 1);
        gemm(pre_w2[0], pre_b2[0], bigA, bigB, cout, cout, GSk, Lk);
        bn(bigB, bigC, cout, GSk, nGk, 1);
        // block 1: bigB -> bigA -> bigC (+res bigB)
        gemm(pre_w1[1], pre_b1[1], bigB, bigA, cout, cout, GSk, Lk);
        bn(bigA, nullptr, cout, GSk, nGk, 1);
        gemm(pre_w2[1], pre_b2[1], bigA, bigC, cout, cout, GSk, Lk);
        bn(bigC, bigB, cout, GSk, nGk, 1);
        // max over K -> h1 [B,cout,S]
        long long th = (long long)BATCH * cout * S;
        maxk_kernel<<<gsz(th), 256, 0, stream>>>(bigC, h1, S, cout, th);
        // PosExtraction: 2 residual blocks on [B, C, S]
        const long long GSs = S;
        const long long Ls  = (long long)BATCH * S;
        gemm(pos_w1[0], pos_b1[0], h1, s0, cout, cout, GSs, Ls);
        bn(s0, nullptr, cout, GSs, BATCH, 1);
        gemm(pos_w2[0], pos_b2[0], s0, s1, cout, cout, GSs, Ls);
        bn(s1, h1, cout, GSs, BATCH, 1);
        gemm(pos_w1[1], pos_b1[1], s1, s0, cout, cout, GSs, Ls);
        bn(s0, nullptr, cout, GSs, BATCH, 1);
        gemm(pos_w2[1], pos_b2[1], s0, h0, cout, cout, GSs, Ls);
        bn(h0, s1, cout, GSs, BATCH, 1);
        // advance
        float* t = curXyz; curXyz = newXyz; newXyz = t;
        Np = S;
    }

    // ---- global max pool + classifier ----
    long long tf = (long long)BATCH * 1024;
    maxs_kernel<<<gsz(tf), 256, 0, stream>>>(h0, feat, 1024, ANC[3], tf);
    gemm(cls_w1, cls_b1, feat, fc1, 512, 1024, 1, BATCH);
    bn(fc1, nullptr, 512, 1, BATCH, 1);
    gemm(cls_w2, cls_b2, fc1, fc2, 256, 512, 1, BATCH);
    bn(fc2, nullptr, 256, 1, BATCH, 1);
    gemm(cls_w3, cls_b3, fc2, (float*)d_out, 120, 256, 1, BATCH);
}
